// CuriosityModule_55027120996868
// MI455X (gfx1250) — matrix operand order repaired and checked by reading, not compiled, not run on hardware
//
#include <hip/hip_runtime.h>
#include <hip/hip_bf16.h>

#define D_DIM 128
#define A_DIM 32
#define M_PAD 112   // 100 seen rows padded to 7 tiles of 16

typedef __attribute__((ext_vector_type(16))) _Float16 v16h;
typedef __attribute__((ext_vector_type(8)))  _Float16 h8v;
typedef __attribute__((ext_vector_type(8)))  float    v8f;

// ---- A-fragment (16x32, f16) per ISA layout:
// lanes 0-15: M=lane, elems 0..7 = K 0..7, elems 8..15 = K 16..23
// lanes 16-31: M=lane-16, elems 0..7 = K 8..15, elems 8..15 = K 24..31
__device__ __forceinline__ v16h load_afrag_f32(const float* __restrict__ X, int ld, int kc) {
    const int lane  = threadIdx.x & 31;
    const int m     = lane & 15;
    const int kbase = kc * 32 + ((lane >> 4) << 3);
    const float* p  = X + (long long)m * ld + kbase;
    union { v16h v; _Float16 e[16]; } u;
#pragma unroll
    for (int i = 0; i < 8; ++i) u.e[i]     = (_Float16)p[i];
#pragma unroll
    for (int i = 0; i < 8; ++i) u.e[8 + i] = (_Float16)p[16 + i];
    return u.v;
}

__device__ __forceinline__ v16h load_afrag_f16(const _Float16* __restrict__ H, int kc) {
    const int lane  = threadIdx.x & 31;
    const int m     = lane & 15;
    const int kbase = kc * 32 + ((lane >> 4) << 3);
    const _Float16* p = H + m * D_DIM + kbase;
    union { v16h v; h8v h[2]; } u;
    u.h[0] = *(const h8v*)(p);
    u.h[1] = *(const h8v*)(p + 16);
    return u.v;
}

// ---- B-fragment (32x16, f16): lanes 0-15: N=lane, elems 0..7 = K 0..7, elems 8..15 = K 16..23;
// lanes 16-31: N=lane-16, elems 0..7 = K 8..15, elems 8..15 = K 24..31.  B = W^T so column N = row of W.
__device__ __forceinline__ v16h load_bfrag(const _Float16* __restrict__ W, int row, int ldk, int kc) {
    const int lane  = threadIdx.x & 31;
    const int kbase = kc * 32 + ((lane >> 4) << 3);
    const _Float16* p = W + (long long)row * ldk + kbase;
    union { v16h v; h8v h[2]; } u;
    u.h[0] = *(const h8v*)(p);
    u.h[1] = *(const h8v*)(p + 16);
    return u.v;
}

#define WMMA(a, b, c) __builtin_amdgcn_wmma_f32_16x16x32_f16(false, (a), false, (b), (short)0, (c), false, false)

// ---------------- Prep: f32 -> f16 weights, seen padding, ||seen||^2 ----------------
__global__ __launch_bounds__(256) void curiosity_prep(
    const float* __restrict__ W1, const float* __restrict__ W2,
    const float* __restrict__ Wf, const float* __restrict__ seen,
    _Float16* __restrict__ W1h, _Float16* __restrict__ W2h,
    _Float16* __restrict__ Wfh, _Float16* __restrict__ seenh,
    float* __restrict__ s2)
{
    const int tid = blockIdx.x * blockDim.x + threadIdx.x;
    if (tid < D_DIM * (D_DIM + A_DIM)) W1h[tid] = (_Float16)W1[tid];
    if (tid < D_DIM * D_DIM) {
        W2h[tid] = (_Float16)W2[tid];
        Wfh[tid] = (_Float16)Wf[tid];
    }
    if (tid < M_PAD * D_DIM) {
        const int j = tid >> 7;
        seenh[tid] = (j < 100) ? (_Float16)seen[tid] : (_Float16)0.0f;
    }
    if (tid < M_PAD) {
        float s = 1e30f;
        if (tid < 100) {
            s = 0.0f;
            for (int k = 0; k < D_DIM; ++k) { float v = seen[tid * D_DIM + k]; s += v * v; }
        }
        s2[tid] = s;
    }
}

// ---------------- Main: full pipeline, one wave per 16 batch rows ----------------
__global__ __launch_bounds__(256) void curiosity_main(
    const float* __restrict__ state, const float* __restrict__ action,
    const float* __restrict__ next_state,
    const float* __restrict__ b1, const float* __restrict__ b2,
    const float* __restrict__ bf,
    const _Float16* __restrict__ W1h, const _Float16* __restrict__ W2h,
    const _Float16* __restrict__ Wfh, const _Float16* __restrict__ seenh,
    const float* __restrict__ s2, float* __restrict__ out, int Btot)
{
    __shared__ __align__(32) _Float16 smem[8][16 * D_DIM];   // 4 KB per wave
    const int lane    = threadIdx.x & 31;
    const int wave    = threadIdx.x >> 5;
    const int n_lo    = lane & 15;
    const int half_id = lane >> 4;        // C layout: rows 0-7 vs 8-15
    const long long b0 = (long long)blockIdx.x * 128 + wave * 16;
    _Float16* hs = smem[wave];

    // ===== Stage 1: h = relu([state,action] @ W1^T + b1), K = 160 = 5 chunks
    v16h ax[5];
#pragma unroll
    for (int kc = 0; kc < 4; ++kc) ax[kc] = load_afrag_f32(state + b0 * D_DIM, D_DIM, kc);
    ax[4] = load_afrag_f32(action + b0 * A_DIM, A_DIM, 0);

#pragma unroll
    for (int t = 0; t < 8; ++t) {
        v8f acc = {};
#pragma unroll
        for (int kc = 0; kc < 5; ++kc)
            acc = WMMA(ax[kc], load_bfrag(W1h, t * 16 + n_lo, D_DIM + A_DIM, kc), acc);
        const float bias = b1[t * 16 + n_lo];
#pragma unroll
        for (int i = 0; i < 8; ++i) {
            float v = acc[i] + bias;
            hs[(half_id * 8 + i) * D_DIM + t * 16 + n_lo] = (_Float16)(v > 0.f ? v : 0.f);
        }
    }
    __syncthreads();

    // ===== Stage 2: pred = h @ W2^T + b2
    v16h ah[4];
#pragma unroll
    for (int kc = 0; kc < 4; ++kc) ah[kc] = load_afrag_f16(hs, kc);
    __syncthreads();
#pragma unroll
    for (int t = 0; t < 8; ++t) {
        v8f acc = {};
#pragma unroll
        for (int kc = 0; kc < 4; ++kc)
            acc = WMMA(ah[kc], load_bfrag(W2h, t * 16 + n_lo, D_DIM, kc), acc);
        const float bias = b2[t * 16 + n_lo];
#pragma unroll
        for (int i = 0; i < 8; ++i)
            hs[(half_id * 8 + i) * D_DIM + t * 16 + n_lo] = (_Float16)(acc[i] + bias);
    }
    __syncthreads();

    // ===== Stage 3: features_pred / features_actual through Wf, accumulate stats
    v16h ap[4], an[4];
#pragma unroll
    for (int kc = 0; kc < 4; ++kc) ap[kc] = load_afrag_f16(hs, kc);
#pragma unroll
    for (int kc = 0; kc < 4; ++kc) an[kc] = load_afrag_f32(next_state + b0 * D_DIM, D_DIM, kc);
    __syncthreads();

    float err[8], a2v[8];
#pragma unroll
    for (int i = 0; i < 8; ++i) { err[i] = 0.f; a2v[i] = 0.f; }

#pragma unroll
    for (int t = 0; t < 8; ++t) {
        v8f fp = {}, fa = {};
#pragma unroll
        for (int kc = 0; kc < 4; ++kc) {
            v16h b = load_bfrag(Wfh, t * 16 + n_lo, D_DIM, kc);
            fp = WMMA(ap[kc], b, fp);
            fa = WMMA(an[kc], b, fa);
        }
        const float bias = bf[t * 16 + n_lo];
#pragma unroll
        for (int i = 0; i < 8; ++i) {
            float vp = fmaxf(fp[i] + bias, 0.f);
            float va = fmaxf(fa[i] + bias, 0.f);
            float d  = vp - va;
            err[i] += d * d;
            a2v[i] += va * va;
            hs[(half_id * 8 + i) * D_DIM + t * 16 + n_lo] = (_Float16)va;
        }
    }
    __syncthreads();

    // butterfly-reduce err / ||fa||^2 across the 16 column-lanes of each half
#pragma unroll
    for (int m = 1; m < 16; m <<= 1) {
#pragma unroll
        for (int i = 0; i < 8; ++i) {
            err[i] += __shfl_xor(err[i], m, 32);
            a2v[i] += __shfl_xor(a2v[i], m, 32);
        }
    }

    // ===== Stage 4: 1-NN over seen (padded to 112 = 7 tiles)
    v16h af[4];
#pragma unroll
    for (int kc = 0; kc < 4; ++kc) af[kc] = load_afrag_f16(hs, kc);

    float mind[8];
#pragma unroll
    for (int i = 0; i < 8; ++i) mind[i] = 3.4e38f;

#pragma unroll
    for (int t = 0; t < 7; ++t) {
        v8f dot = {};
#pragma unroll
        for (int kc = 0; kc < 4; ++kc)
            dot = WMMA(af[kc], load_bfrag(seenh, t * 16 + n_lo, D_DIM, kc), dot);
        const float s2j = s2[t * 16 + n_lo];
#pragma unroll
        for (int i = 0; i < 8; ++i) {
            float d2 = a2v[i] + s2j - 2.f * dot[i];
            mind[i] = fminf(mind[i], d2);
        }
    }
#pragma unroll
    for (int m = 1; m < 16; m <<= 1)
#pragma unroll
        for (int i = 0; i < 8; ++i)
            mind[i] = fminf(mind[i], __shfl_xor(mind[i], m, 32));

    // ===== Output [3, B]
    if (n_lo == 0) {
        const long long base = b0 + half_id * 8;
#pragma unroll
        for (int i = 0; i < 8; ++i) {
            float pe = err[i] * (1.0f / 128.0f);
            float nv = fminf(1.0f, sqrtf(fmaxf(mind[i], 0.0f)) * 0.1f);
            out[base + i]                       = pe;
            out[(long long)Btot + base + i]     = nv;
            out[2LL * Btot + base + i]          = 0.5f * pe + 0.5f * nv;
        }
    }
}

extern "C" void kernel_launch(void* const* d_in, const int* in_sizes, int n_in,
                              void* d_out, int out_size, void* d_ws, size_t ws_size,
                              hipStream_t stream) {
    const float* state      = (const float*)d_in[0];
    const float* action     = (const float*)d_in[1];
    const float* next_state = (const float*)d_in[2];
    const float* seen       = (const float*)d_in[3];
    const float* W1         = (const float*)d_in[4];
    const float* b1         = (const float*)d_in[5];
    const float* W2         = (const float*)d_in[6];
    const float* b2         = (const float*)d_in[7];
    const float* Wf         = (const float*)d_in[8];
    const float* bf         = (const float*)d_in[9];
    float* out              = (float*)d_out;

    const int B = in_sizes[0] / D_DIM;   // 262144

    // Workspace carve-up (f16 weights + seen-norms), all 256B-aligned chunks
    char* ws = (char*)d_ws;
    _Float16* W1h   = (_Float16*)ws; ws += (size_t)D_DIM * (D_DIM + A_DIM) * 2;
    _Float16* W2h   = (_Float16*)ws; ws += (size_t)D_DIM * D_DIM * 2;
    _Float16* Wfh   = (_Float16*)ws; ws += (size_t)D_DIM * D_DIM * 2;
    _Float16* seenh = (_Float16*)ws; ws += (size_t)M_PAD * D_DIM * 2;
    float*    s2    = (float*)ws;

    const int prepElems = D_DIM * (D_DIM + A_DIM);           // 20480 (largest)
    curiosity_prep<<<(prepElems + 255) / 256, 256, 0, stream>>>(
        W1, W2, Wf, seen, W1h, W2h, Wfh, seenh, s2);

    curiosity_main<<<B / 128, 256, 0, stream>>>(
        state, action, next_state, b1, b2, bf,
        W1h, W2h, Wfh, seenh, s2, out, B);
}
